// SelectiveScan_33122787787294
// MI455X (gfx1250) — matrix operand order repaired and checked by reading, not compiled, not run on hardware
//
#include <hip/hip_runtime.h>
#include <cmath>

typedef __attribute__((ext_vector_type(16))) __bf16 v16bf;
typedef __attribute__((ext_vector_type(8)))  __bf16 v8bf;
typedef __attribute__((ext_vector_type(8)))  float  v8f;

#define BSZ    2
#define LSEQ   1024
#define DMODEL 1024
#define DINNER 2048
#define DSTATE 16
#define DCONV  4
#define NROWS  (BSZ * LSEQ)         // 2048 token rows

// GEMM tiling
#define TM  128                     // block tile M
#define TN  128                     // block tile N
#define TK  64                      // K step per barrier interval
#define LDT 72                      // padded LDS row stride (bf16 elems; 144B = 9*16B)

// Async global->LDS 16B copy (gfx1250): per active lane, LDS[vdst] <= MEM[vaddr].
// Tracked by ASYNCcnt. Low 32 bits of a flat LDS pointer are the LDS byte
// address (ISA 10.2).
#define ASYNC_COPY_B128(ldsPtr, gPtr)                                         \
    asm volatile("global_load_async_to_lds_b128 %0, %1, off"                  \
                 :: "v"((unsigned)(uintptr_t)(ldsPtr)), "v"(gPtr) : "memory")
#define WAIT_ASYNC0() asm volatile("s_wait_asynccnt 0" ::: "memory")

// ---------------------------------------------------------------------------
// fp32 -> bf16 (no transpose), for the activation operand x
// ---------------------------------------------------------------------------
__global__ void cvt_f32_bf16(const float* __restrict__ in,
                             __bf16* __restrict__ out, int n) {
    int i = blockIdx.x * 256 + threadIdx.x;
    if (i < n) out[i] = (__bf16)in[i];
}

// ---------------------------------------------------------------------------
// fp32 -> bf16 with transpose: in[R][C] -> out[C][R].
// Weights become N-major so the WMMA B-operand fragment (lane n holds
// K-contiguous values) is loadable with aligned ds_load_b128 — same as A.
// R, C multiples of 32.
// ---------------------------------------------------------------------------
__global__ __launch_bounds__(256)
void cvt_transpose_bf16(const float* __restrict__ in, __bf16* __restrict__ out,
                        int R, int C) {
    __shared__ float tile[32][33];
    const int c0 = blockIdx.x * 32, r0 = blockIdx.y * 32;
    const int tx = threadIdx.x & 31, ty = threadIdx.x >> 5;   // 32 x 8
    #pragma unroll
    for (int i = 0; i < 32; i += 8)
        tile[ty + i][tx] = in[(size_t)(r0 + ty + i) * C + c0 + tx];
    __syncthreads();
    #pragma unroll
    for (int i = 0; i < 32; i += 8)
        out[(size_t)(c0 + ty + i) * R + r0 + tx] = (__bf16)tile[tx][ty + i];
}

// ---------------------------------------------------------------------------
// Tiled bf16 WMMA GEMM: C[M,N] = A[M,K] * BT[N,K]^T, fp32 accumulate.
// 128 threads (4 waves). Block tile 128x128, wave tile 64x64 = 4x4 fragments
// of v_wmma_f32_16x16x32_bf16; K staged 64 wide -> 32 WMMAs per wave per
// barrier interval. Tiles staged with global_load_async_to_lds_b128.
// EPI==1 fuses delta = softplus(acc + bias[col]).
// ---------------------------------------------------------------------------
template <int EPI>
__global__ __launch_bounds__(128)
void gemm_bf16_wmma(const __bf16* __restrict__ A, const __bf16* __restrict__ BT,
                    float* __restrict__ C, const float* __restrict__ bias,
                    int M, int N, int K) {
    __shared__ __bf16 As[TM * LDT];
    __shared__ __bf16 Bs[TN * LDT];

    const int tid  = threadIdx.x;
    const int lane = tid & 31;
    const int wave = tid >> 5;
    const int wm   = (wave >> 1) * 64;       // wave tile origin in block tile
    const int wn   = (wave & 1) * 64;
    const int rowBase = blockIdx.y * TM;
    const int colBase = blockIdx.x * TN;
    const int m16 = lane & 15;               // M/N index within a fragment
    const int kh  = lane >> 4;               // K-half select (ISA 16-bit layout)

    v8f acc[4][4];
    #pragma unroll
    for (int i = 0; i < 4; ++i)
        #pragma unroll
        for (int j = 0; j < 4; ++j) acc[i][j] = v8f{};

    for (int k0 = 0; k0 < K; k0 += TK) {
        // Stage A tile 128x64 and BT tile 128x64: 8 async 16B copies each
        // per thread; lanes' addresses are 16B aligned and K-contiguous.
        #pragma unroll
        for (int i = 0; i < 8; ++i) {
            int e = i * 128 + tid;           // chunk id
            int r = e >> 3, ch = e & 7;      // 8 chunks of 8 bf16 per row
            ASYNC_COPY_B128(&As[r * LDT + ch * 8],
                            A + (size_t)(rowBase + r) * K + k0 + ch * 8);
        }
        #pragma unroll
        for (int i = 0; i < 8; ++i) {
            int e = i * 128 + tid;
            int r = e >> 3, ch = e & 7;
            ASYNC_COPY_B128(&Bs[r * LDT + ch * 8],
                            BT + (size_t)(colBase + r) * K + k0 + ch * 8);
        }
        WAIT_ASYNC0();
        __syncthreads();

        #pragma unroll
        for (int kk = 0; kk < TK; kk += 32) {
            // A fragment (16-bit A 16x32): lanes 0-15 K 0..7 / 16..23,
            // lanes 16-31 K 8..15 / 24..31 -> two aligned b128 LDS loads.
            v16bf af[4], bfr[4];
            #pragma unroll
            for (int t = 0; t < 4; ++t) {
                const __bf16* ap = &As[(wm + t * 16 + m16) * LDT + kk];
                v8bf alo = *(const v8bf*)(ap + kh * 8);
                v8bf ahi = *(const v8bf*)(ap + 16 + kh * 8);
                af[t] = __builtin_shufflevector(alo, ahi, 0, 1, 2, 3, 4, 5, 6, 7,
                                                8, 9, 10, 11, 12, 13, 14, 15);
                // B fragment: same K order as A: lanes 0-15 K 0..7 / 16..23,
                // lanes 16-31 K 8..15 / 24..31; BT is N-major so contiguous.
                const __bf16* bp = &Bs[(wn + t * 16 + m16) * LDT + kk + kh * 8];
                v8bf blo = *(const v8bf*)(bp);
                v8bf bhi = *(const v8bf*)(bp + 16);
                bfr[t] = __builtin_shufflevector(blo, bhi, 0, 1, 2, 3, 4, 5, 6, 7,
                                                 8, 9, 10, 11, 12, 13, 14, 15);
            }
            #pragma unroll
            for (int i = 0; i < 4; ++i)
                #pragma unroll
                for (int j = 0; j < 4; ++j)
                    acc[i][j] = __builtin_amdgcn_wmma_f32_16x16x32_bf16(
                        false, af[i], false, bfr[j], (short)0, acc[i][j],
                        false, false);
        }
        __syncthreads();
    }

    // Epilogue. C/D layout: element v lives at m = kh*8 + v, n = m16.
    #pragma unroll
    for (int i = 0; i < 4; ++i) {
        #pragma unroll
        for (int v = 0; v < 8; ++v) {
            int r = rowBase + wm + i * 16 + kh * 8 + v;
            #pragma unroll
            for (int j = 0; j < 4; ++j) {
                int cc = colBase + wn + j * 16 + m16;
                float val = acc[i][j][v];
                if (EPI == 1) {              // delta = softplus(acc + b_dt)
                    val += bias[cc];
                    val = (val > 20.f) ? val : log1pf(__expf(val));
                }
                C[(size_t)r * N + cc] = val;
            }
        }
    }
}

// ---------------------------------------------------------------------------
// Depthwise causal conv (K=4) + bias + SiLU. Reads xs = xz[..., :DINNER].
// Writes xc (fp32, for scan/GEMV) and xc_bf16 (for the W_dt WMMA GEMM).
// ---------------------------------------------------------------------------
__global__ __launch_bounds__(256)
void conv_silu(const float* __restrict__ xz, const float* __restrict__ cw,
               const float* __restrict__ cb, float* __restrict__ xc,
               __bf16* __restrict__ xcb) {
    int idx = blockIdx.x * 256 + threadIdx.x;      // (b*L + l)*DINNER + d
    int d = idx % DINNER;
    int l = (idx / DINNER) % LSEQ;
    int b = idx / (DINNER * LSEQ);
    float u = cb[d];
    #pragma unroll
    for (int j = 0; j < DCONV; ++j) {
        int ll = l - (DCONV - 1) + j;
        if (ll >= 0)
            u += cw[d * DCONV + j] *
                 xz[(size_t)(b * LSEQ + ll) * (2 * DINNER) + d];
    }
    float s = u / (1.f + __expf(-u));              // SiLU
    xc[idx]  = s;
    xcb[idx] = (__bf16)s;
}

// ---------------------------------------------------------------------------
// B_ssm[row, s] = sum_k xc[row,k] * W_x[k, DSTATE+s]
// (only the B half of x_dbl is ever consumed by the reference)
// ---------------------------------------------------------------------------
__global__ __launch_bounds__(256)
void gemv_bssm(const float* __restrict__ xc, const float* __restrict__ Wx,
               float* __restrict__ Bssm) {
    int idx = blockIdx.x * 256 + threadIdx.x;      // NROWS*DSTATE threads
    int s   = idx & (DSTATE - 1);
    int row = idx / DSTATE;
    const float* xr = xc + (size_t)row * DINNER;
    float acc = 0.f;
    for (int k = 0; k < DINNER; ++k)
        acc += xr[k] * Wx[k * (2 * DSTATE) + DSTATE + s];
    Bssm[idx] = acc;
}

// ---------------------------------------------------------------------------
// Selective scan. One thread per (b, d) channel; h[16] in registers;
// B_ssm staged per 64-step chunk in LDS. Fuses +xc*Dp and *silu(z),
// emits bf16 directly for the output GEMM.
// ---------------------------------------------------------------------------
__global__ __launch_bounds__(256)
void scan_kernel(const float* __restrict__ delta, const float* __restrict__ xc,
                 const float* __restrict__ Bssm, const float* __restrict__ xz,
                 const float* __restrict__ A_log, const float* __restrict__ Dp,
                 __bf16* __restrict__ ybf) {
    __shared__ float Bs[64][DSTATE];
    const int tid = threadIdx.x;
    const int d   = blockIdx.x * 256 + tid;
    const int b   = blockIdx.y;

    float Arow[DSTATE], h[DSTATE];
    #pragma unroll
    for (int s = 0; s < DSTATE; ++s) {
        Arow[s] = -__expf(A_log[d * DSTATE + s]);  // A = -exp(A_log)
        h[s] = 0.f;
    }
    const float dpd = Dp[d];

    for (int l0 = 0; l0 < LSEQ; l0 += 64) {
        for (int i = tid; i < 64 * DSTATE; i += 256)
            Bs[i >> 4][i & 15] =
                Bssm[((size_t)(b * LSEQ + l0) + (i >> 4)) * DSTATE + (i & 15)];
        __syncthreads();
        for (int li = 0; li < 64; ++li) {
            const size_t rl = (size_t)(b * LSEQ + l0 + li);
            const float dt = delta[rl * DINNER + d];
            const float xv = xc[rl * DINNER + d];
            float y = 0.f;
            #pragma unroll
            for (int s = 0; s < DSTATE; ++s) {
                float dA = __expf(dt * Arow[s]);
                h[s] = dA * h[s] + dt * Bs[li][s] * xv;
                y += h[s];
            }
            y += xv * dpd;
            const float z = xz[rl * (2 * DINNER) + DINNER + d];
            y *= z / (1.f + __expf(-z));           // * silu(z)
            ybf[rl * DINNER + d] = (__bf16)y;
        }
        __syncthreads();
    }
}

// ---------------------------------------------------------------------------
extern "C" void kernel_launch(void* const* d_in, const int* in_sizes, int n_in,
                              void* d_out, int out_size, void* d_ws,
                              size_t ws_size, hipStream_t stream) {
    (void)in_sizes; (void)n_in; (void)out_size; (void)ws_size;
    const float* x     = (const float*)d_in[0];   // [B,L,DMODEL]
    const float* W_in  = (const float*)d_in[1];   // [DMODEL, 2*DINNER]
    const float* convw = (const float*)d_in[2];   // [DINNER,1,4]
    const float* convb = (const float*)d_in[3];   // [DINNER]
    const float* W_x   = (const float*)d_in[4];   // [DINNER, 2*DSTATE]
    const float* W_dt  = (const float*)d_in[5];   // [DINNER, DINNER]
    const float* b_dt  = (const float*)d_in[6];   // [DINNER]
    const float* A_log = (const float*)d_in[7];   // [DINNER, DSTATE]
    const float* Dp    = (const float*)d_in[8];   // [DINNER]
    const float* W_out = (const float*)d_in[9];   // [DINNER, DMODEL]
    float* out = (float*)d_out;                   // [B,L,DMODEL]

    char* ws = (char*)d_ws;
    size_t off = 0;
    auto alloc = [&](size_t bytes) -> void* {
        void* p = ws + off;
        off = (off + bytes + 255) & ~(size_t)255;
        return p;
    };

    __bf16* xbf   = (__bf16*)alloc((size_t)NROWS * DMODEL * 2);
    __bf16* WinT  = (__bf16*)alloc((size_t)DMODEL * 2 * DINNER * 2); // [4096][1024]
    __bf16* WdtT  = (__bf16*)alloc((size_t)DINNER * DINNER * 2);     // [2048][2048]
    __bf16* WoutT = (__bf16*)alloc((size_t)DINNER * DMODEL * 2);     // [1024][2048]
    float*  xz    = (float*) alloc((size_t)NROWS * 2 * DINNER * 4);
    float*  xc    = (float*) alloc((size_t)NROWS * DINNER * 4);
    __bf16* xcb   = (__bf16*)alloc((size_t)NROWS * DINNER * 2);
    float*  delta = (float*) alloc((size_t)NROWS * DINNER * 4);
    float*  Bssm  = (float*) alloc((size_t)NROWS * DSTATE * 4);
    __bf16* ybf   = (__bf16*)alloc((size_t)NROWS * DINNER * 2);

    // 1) bf16 operand prep (weights transposed to N-major for the B operand)
    cvt_f32_bf16<<<(NROWS * DMODEL + 255) / 256, 256, 0, stream>>>(
        x, xbf, NROWS * DMODEL);
    cvt_transpose_bf16<<<dim3(2 * DINNER / 32, DMODEL / 32), 256, 0, stream>>>(
        W_in, WinT, DMODEL, 2 * DINNER);
    cvt_transpose_bf16<<<dim3(DINNER / 32, DINNER / 32), 256, 0, stream>>>(
        W_dt, WdtT, DINNER, DINNER);
    cvt_transpose_bf16<<<dim3(DMODEL / 32, DINNER / 32), 256, 0, stream>>>(
        W_out, WoutT, DINNER, DMODEL);

    // 2) xz = x @ W_in   [2048x1024] * [1024x4096]
    gemm_bf16_wmma<0><<<dim3(2 * DINNER / TN, NROWS / TM), 128, 0, stream>>>(
        xbf, WinT, xz, nullptr, NROWS, 2 * DINNER, DMODEL);

    // 3) xc = silu(conv1d(xs) + conv_b)
    conv_silu<<<(NROWS * DINNER) / 256, 256, 0, stream>>>(
        xz, convw, convb, xc, xcb);

    // 4) delta = softplus(xc @ W_dt + b_dt)   [2048x2048]*[2048x2048]
    gemm_bf16_wmma<1><<<dim3(DINNER / TN, NROWS / TM), 128, 0, stream>>>(
        xcb, WdtT, delta, b_dt, NROWS, DINNER, DINNER);

    // 5) B_ssm = xc @ W_x[:, DSTATE:]
    gemv_bssm<<<(NROWS * DSTATE) / 256, 256, 0, stream>>>(xc, W_x, Bssm);

    // 6) selective scan (+ Dp skip, * silu(z)), writes bf16
    scan_kernel<<<dim3(DINNER / 256, BSZ), 256, 0, stream>>>(
        delta, xc, Bssm, xz, A_log, Dp, ybf);

    // 7) out = y @ W_out   [2048x2048]*[2048x1024]
    gemm_bf16_wmma<0><<<dim3(DMODEL / TN, NROWS / TM), 128, 0, stream>>>(
        ybf, WoutT, out, nullptr, NROWS, DMODEL, DINNER);
}
